// LSTM_77043123356084
// MI455X (gfx1250) — matrix operand order repaired and checked by reading, not compiled, not run on hardware
//
#include <hip/hip_runtime.h>
#include <hip/hip_bf16.h>
#include <stdint.h>

// ---------------------------------------------------------------------------
// Types for WMMA operands (gfx1250, wave32)
// ---------------------------------------------------------------------------
typedef __bf16 bf16;
typedef __attribute__((ext_vector_type(16))) __bf16 v16bf;
typedef __attribute__((ext_vector_type(8)))  float  v8f;
typedef __attribute__((ext_vector_type(4)))  unsigned int u32x4;

union VA {
    v16bf v;
    u32x4 u[2];
};

#define BATCH 32
#define SEQ 2048
#define IN0 256
#define HID 512
#define G4 2048          // 4*HID
#define OUT 256
#define REC_WGS 32       // workgroups in recurrence grid (16 hidden units each)
#define LDS_BPAD 72      // 64 B-cols + 8 pad (bf16 elems) per K row
#define LDS_GPAD 68      // 64 gate cols + 4 pad (f32) per batch row

__device__ __forceinline__ float sigf(float x) {
    return 1.0f / (1.0f + __expf(-x));
}

// Global barrier across the 32 recurrence workgroups (monotonic counter).
__device__ __forceinline__ void grid_sync(int* cnt, int target) {
    __syncthreads();
    if (threadIdx.x == 0) {
        __threadfence();                 // release h writes
        atomicAdd(cnt, 1);
        volatile int* vc = (volatile int*)cnt;
        while (*vc < target) { __builtin_amdgcn_s_sleep(1); }
    }
    __syncthreads();
    __threadfence();                     // acquire other WGs' h writes
}

// ---------------------------------------------------------------------------
// Prep kernels: fp32 -> bf16 conversions / weight transposition
// ---------------------------------------------------------------------------
__global__ void cvt_x(const float* __restrict__ x, bf16* __restrict__ xb, int n) {
    int i = blockIdx.x * 256 + threadIdx.x;
    if (i < n) xb[i] = (bf16)x[i];
}

// lw[k][g] (K x 4H, bf16): rows 0..Kx-1 = w_ih^T, rows Kx..K-1 = w_hh^T
__global__ void build_lw(const float* __restrict__ wx, const float* __restrict__ wh,
                         bf16* __restrict__ lw, int Kx, int K) {
    int i = blockIdx.x * 256 + threadIdx.x;
    if (i >= K * G4) return;
    int k = i >> 11;          // / 2048
    int g = i & (G4 - 1);
    float v = (k < Kx) ? wx[(size_t)g * Kx + k] : wh[(size_t)g * HID + (k - Kx)];
    lw[i] = (bf16)v;
}

// wfcT[k][o] (512 x 256, bf16) = w_fc[o][k]
__global__ void build_wfcT(const float* __restrict__ wfc, bf16* __restrict__ wt) {
    int i = blockIdx.x * 256 + threadIdx.x;  // < 512*256
    int k = i >> 8;
    int o = i & 255;
    wt[i] = (bf16)wfc[(size_t)o * HID + k];
}

// ---------------------------------------------------------------------------
// LSTM recurrence: one layer, sequential over T, 32 WGs x 256 threads.
// Each WG owns 16 hidden units -> 64 gate columns (i,f,g,o x 16).
// Weights (K x 64 slice) resident in LDS for all 2048 steps.
// K, Kx, LAYER are compile-time so all operand addresses are base+imm.
// ---------------------------------------------------------------------------
#define LOAD_A(dst, p)                                              \
    {                                                               \
        (dst).u[0] = *(const u32x4*)((p) + half * 8);               \
        (dst).u[1] = *(const u32x4*)((p) + 16 + half * 8);          \
    }
#define LOAD_B(dst, kb)                                             \
    {                                                               \
        const bf16* bp_ = bw + (size_t)(kb) * LDS_BPAD + nt * 16 + (lane & 15); \
        _Pragma("unroll") for (int s_ = 0; s_ < 16; ++s_)          \
            (dst).v[s_] = bp_[(size_t)(half * 8 + (s_ & 7) + (s_ >> 3) * 16) * LDS_BPAD]; \
    }

template <int K, int Kx, int LAYER>
__global__ void lstm_rec(const bf16* __restrict__ xsrc, const bf16* __restrict__ lw,
                         const float* __restrict__ bias, bf16* __restrict__ hbuf,
                         bf16* __restrict__ hseq, int* cnt) {
    extern __shared__ char smem[];
    bf16*  bw = (bf16*)smem;                                   // [K][72] bf16
    float* gl = (float*)(smem + (size_t)K * LDS_BPAD * 2);     // [32][68] f32

    const int tid  = threadIdx.x;
    const int wg   = blockIdx.x;
    const int lane = tid & 31;
    const int wave = tid >> 5;     // 0..7
    const int mt   = wave & 1;     // batch half (rows 0-15 / 16-31)
    const int nt   = wave >> 1;    // gate index 0..3 (i,f,g,o)
    const int half = lane >> 4;
    const int mrow = lane & 15;
    const int brow = mt * 16 + mrow;

    // ---- load this WG's weight slice into LDS via async global->LDS copy ----
    {
        const int chunks = K * 8;  // 16B chunks: 64 cols * 2B / 16B = 8 per K row
        uint32_t ldsbase = (uint32_t)(uintptr_t)bw;
        for (int c = tid; c < chunks; c += 256) {
            int k  = c >> 3;
            int q  = c & 7;
            int g  = q >> 1;            // gate
            int j0 = (q & 1) * 8;       // 8 bf16 within the 16-col gate slice
            const bf16* src = lw + (size_t)k * G4 + g * HID + wg * 16 + j0;
            uint32_t dst = ldsbase + (uint32_t)(k * (LDS_BPAD * 2) + q * 16);
            asm volatile("global_load_async_to_lds_b128 %0, %1, off"
                         :: "v"(dst), "v"(src) : "memory");
        }
        asm volatile("s_wait_asynccnt 0" ::: "memory");
    }
    __syncthreads();

    // ---- zero h state (own slice) + cell state in registers ----
    float cst[2] = {0.f, 0.f};
    #pragma unroll
    for (int p = 0; p < 2; ++p) {
        int idx = tid * 2 + p;          // 0..511
        int b = idx >> 4, j = idx & 15;
        hbuf[b * HID + wg * 16 + j] = (bf16)0.f;
    }
    int target = REC_WGS;
    grid_sync(cnt, target); target += REC_WGS;

    const float bv = bias[nt * HID + wg * 16 + (lane & 15)];  // per-column bias
    const bf16* hrow = hbuf + (size_t)brow * HID;
    constexpr int NX = Kx / 32;
    constexpr int NH = (K - Kx) / 32;

    for (int t = 0; t < SEQ; ++t) {
        // A row base: concat(x_t[brow], h[brow]) along K
        const bf16* xrow = xsrc + (LAYER == 0 ? ((size_t)brow * SEQ + t) * IN0
                                              : ((size_t)t * BATCH + brow) * HID);
        // prefetch next timestep's x row (global_prefetch_b8)
        if (t + 1 < SEQ) {
            const bf16* xnext = xsrc + (LAYER == 0 ? ((size_t)brow * SEQ + t + 1) * IN0
                                                   : ((size_t)(t + 1) * BATCH + brow) * HID);
            __builtin_prefetch((const void*)xnext, 0, 1);
        }

        v8f acc = {};
        VA a0, b0, a1, b1;
        LOAD_A(a0, xrow);
        LOAD_B(b0, 0);
        // x-part of K, software-pipelined; boundary tile feeds first h tile
        #pragma unroll
        for (int kc = 0; kc < NX; ++kc) {
            if (kc + 1 < NX) {
                LOAD_A(a1, xrow + (kc + 1) * 32);
            } else {
                LOAD_A(a1, hrow);
            }
            LOAD_B(b1, (kc + 1) * 32);
            acc = __builtin_amdgcn_wmma_f32_16x16x32_bf16(
                      false, a0.v, false, b0.v, (short)0, acc, false, false);
            a0 = a1; b0 = b1;
        }
        // h-part of K
        #pragma unroll
        for (int kc = 0; kc < NH; ++kc) {
            if (kc + 1 < NH) {
                LOAD_A(a1, hrow + (kc + 1) * 32);
                LOAD_B(b1, Kx + (kc + 1) * 32);
            }
            acc = __builtin_amdgcn_wmma_f32_16x16x32_bf16(
                      false, a0.v, false, b0.v, (short)0, acc, false, false);
            a0 = a1; b0 = b1;
        }

        // stash pre-activation gates (+bias) in LDS: gl[b][gate*16+j]
        #pragma unroll
        for (int r = 0; r < 8; ++r) {
            int M  = r + 8 * half;
            int b  = mt * 16 + M;
            int lc = nt * 16 + (lane & 15);
            gl[b * LDS_GPAD + lc] = acc[r] + bv;
        }
        __syncthreads();

        // elementwise LSTM cell update for this WG's 16 hidden units x 32 batch
        #pragma unroll
        for (int p = 0; p < 2; ++p) {
            int idx = tid * 2 + p;          // 0..511
            int b = idx >> 4, j = idx & 15;
            float ig = gl[b * LDS_GPAD + j];
            float fg = gl[b * LDS_GPAD + 16 + j];
            float gg = gl[b * LDS_GPAD + 32 + j];
            float og = gl[b * LDS_GPAD + 48 + j];
            float cn = sigf(fg) * cst[p] + sigf(ig) * tanhf(gg);
            cst[p] = cn;
            float hv = sigf(og) * tanhf(cn);
            bf16 hb = (bf16)hv;
            hbuf[b * HID + wg * 16 + j] = hb;
            hseq[((size_t)t * BATCH + b) * HID + wg * 16 + j] = hb;
        }
        grid_sync(cnt, target); target += REC_WGS;
    }
}

// ---------------------------------------------------------------------------
// Final FC: out[b][t][o] = h1[t*32+b][:] . w_fc[o][:] + b_fc[o]
// M = 65536 rows (t,b), N = 256, K = 512. One 16x16 tile per wave.
// ---------------------------------------------------------------------------
__global__ void fc_kernel(const bf16* __restrict__ h1, const bf16* __restrict__ wfcT,
                          const float* __restrict__ bfc, float* __restrict__ out) {
    const int tid   = threadIdx.x;
    const int lane  = tid & 31;
    const int wave  = tid >> 5;
    const int ntile = blockIdx.y * 8 + wave;  // 0..15
    const int m     = blockIdx.x;             // 0..4095
    const int half  = lane >> 4;
    const int mrow  = lane & 15;

    const bf16* arow = h1 + ((size_t)m * 16 + mrow) * HID;
    float bv = bfc[ntile * 16 + (lane & 15)];
    v8f acc = {bv, bv, bv, bv, bv, bv, bv, bv};

    constexpr int NK = HID / 32;   // 16
    VA a0, b0, a1, b1;
    a0.u[0] = *(const u32x4*)(arow + half * 8);
    a0.u[1] = *(const u32x4*)(arow + 16 + half * 8);
    {
        const bf16* bp = wfcT + ntile * 16 + (lane & 15);
        #pragma unroll
        for (int s = 0; s < 16; ++s)
            b0.v[s] = bp[(size_t)(half * 8 + (s & 7) + (s >> 3) * 16) * OUT];
    }
    #pragma unroll
    for (int kc = 0; kc < NK; ++kc) {
        if (kc + 1 < NK) {
            int kb = (kc + 1) * 32;
            a1.u[0] = *(const u32x4*)(arow + kb + half * 8);
            a1.u[1] = *(const u32x4*)(arow + kb + 16 + half * 8);
            const bf16* bp = wfcT + (size_t)kb * OUT + ntile * 16 + (lane & 15);
            #pragma unroll
            for (int s = 0; s < 16; ++s)
                b1.v[s] = bp[(size_t)(half * 8 + (s & 7) + (s >> 3) * 16) * OUT];
        }
        acc = __builtin_amdgcn_wmma_f32_16x16x32_bf16(
                  false, a0.v, false, b0.v, (short)0, acc, false, false);
        a0 = a1; b0 = b1;
    }

    #pragma unroll
    for (int r = 0; r < 8; ++r) {
        int M   = r + 8 * half;
        int row = m * 16 + M;              // = t*32 + b
        int t   = row >> 5;
        int b   = row & 31;
        out[((size_t)b * SEQ + t) * OUT + ntile * 16 + (lane & 15)] = acc[r];
    }
}

// ---------------------------------------------------------------------------
// Launch
// ---------------------------------------------------------------------------
extern "C" void kernel_launch(void* const* d_in, const int* in_sizes, int n_in,
                              void* d_out, int out_size, void* d_ws, size_t ws_size,
                              hipStream_t stream) {
    (void)in_sizes; (void)n_in; (void)out_size; (void)ws_size;
    const float* x     = (const float*)d_in[0];
    const float* w_ih0 = (const float*)d_in[1];
    const float* w_hh0 = (const float*)d_in[2];
    const float* b0    = (const float*)d_in[3];
    const float* w_ih1 = (const float*)d_in[4];
    const float* w_hh1 = (const float*)d_in[5];
    const float* b1    = (const float*)d_in[6];
    const float* w_fc  = (const float*)d_in[7];
    const float* b_fc  = (const float*)d_in[8];
    float* out = (float*)d_out;

    char* ws = (char*)d_ws;
    size_t off = 0;
    int*  cnt   = (int*)(ws);            off += 256;
    bf16* xbf   = (bf16*)(ws + off);     off += (size_t)BATCH * SEQ * IN0 * 2;
    bf16* lw0   = (bf16*)(ws + off);     off += (size_t)(IN0 + HID) * G4 * 2;
    bf16* lw1   = (bf16*)(ws + off);     off += (size_t)(HID + HID) * G4 * 2;
    bf16* wfcT  = (bf16*)(ws + off);     off += (size_t)HID * OUT * 2;
    bf16* hbuf0 = (bf16*)(ws + off);     off += (size_t)BATCH * HID * 2;
    bf16* hbuf1 = (bf16*)(ws + off);     off += (size_t)BATCH * HID * 2;
    bf16* h0    = (bf16*)(ws + off);     off += (size_t)SEQ * BATCH * HID * 2;
    bf16* h1    = (bf16*)(ws + off);     off += (size_t)SEQ * BATCH * HID * 2;

    hipMemsetAsync(cnt, 0, 256, stream);

    cvt_x<<<(BATCH * SEQ * IN0) / 256, 256, 0, stream>>>(x, xbf, BATCH * SEQ * IN0);
    build_lw<<<((IN0 + HID) * G4) / 256, 256, 0, stream>>>(w_ih0, w_hh0, lw0, IN0, IN0 + HID);
    build_lw<<<((HID + HID) * G4) / 256, 256, 0, stream>>>(w_ih1, w_hh1, lw1, HID, HID + HID);
    build_wfcT<<<(HID * OUT) / 256, 256, 0, stream>>>(w_fc, wfcT);

    size_t smem0 = (size_t)(IN0 + HID) * LDS_BPAD * 2 + 32 * LDS_GPAD * 4;
    size_t smem1 = (size_t)(HID + HID) * LDS_BPAD * 2 + 32 * LDS_GPAD * 4;
    lstm_rec<IN0 + HID, IN0, 0><<<REC_WGS, 256, smem0, stream>>>(xbf, lw0, b0, hbuf0, h0, cnt + 0);
    lstm_rec<HID + HID, HID, 1><<<REC_WGS, 256, smem1, stream>>>(h0,  lw1, b1, hbuf1, h1, cnt + 16);

    fc_kernel<<<dim3((SEQ * BATCH) / 16, 2), 256, 0, stream>>>(h1, wfcT, b_fc, out);
}
